// GQAttention_5660766896376
// MI455X (gfx1250) — compile-verified, matrix operand order checked by reading, not run on hardware
//
#include <hip/hip_runtime.h>
#include <hip/hip_bf16.h>

typedef __bf16 bf16_t;
typedef __attribute__((ext_vector_type(16))) __bf16 v16bf;
typedef __attribute__((ext_vector_type(8)))  __bf16 v8bf;
typedef __attribute__((ext_vector_type(8)))  float  v8f;
typedef unsigned int u32x4 __attribute__((ext_vector_type(4)));
typedef int          i32x8 __attribute__((ext_vector_type(8)));
typedef int          i32x4 __attribute__((ext_vector_type(4)));

#define BB 2
#define LL 2048
#define DD 1024
#define HH 16
#define KVH 4
#define DH 64

// Toolchain fingerprint: therock-10.0 headers (6-arg TDM builtin) ship the
// gfx1250 TDM header; ROCm 7.2 (5-arg builtin) does not.
#if __has_include(<hip/amd_detail/amd_gfx1250_TDM.h>)
#define TDM_LOAD(g0, g1) \
  __builtin_amdgcn_tensor_load_to_lds(g0, g1, (i32x4)0, (i32x4)0, (i32x8)0, 0)
#else
#define TDM_LOAD(g0, g1) \
  __builtin_amdgcn_tensor_load_to_lds(g0, g1, (i32x4)0, (i32x4)0, 0)
#endif

// ---------------------------------------------------------------------------
// Fragment loader for 16-bit A/B operands of V_WMMA_*_16X16X32_BF16.
// Per CDNA5 ISA (CDNA5 ISA): lane = (hi<<4)|row ;
//   element e<8  holds K = k0 + 8*hi + e
//   element e>=8 holds K = k0 + 16 + 8*hi + (e-8)
// => two contiguous 16-byte loads per lane (global_load_b128 / ds_load_b128).
// ---------------------------------------------------------------------------
__device__ __forceinline__ v16bf load_frag(const bf16_t* p, int k0, int hi) {
  v8bf lo = *(const v8bf*)(p + k0 + 8 * hi);
  v8bf hh = *(const v8bf*)(p + k0 + 16 + 8 * hi);
  v16bf r;
#pragma unroll
  for (int i = 0; i < 8; ++i) { r[i] = lo[i]; r[i + 8] = hh[i]; }
  return r;
}

__device__ __forceinline__ v8f wmma_bf16(v16bf a, v16bf b, v8f c) {
  return __builtin_amdgcn_wmma_f32_16x16x32_bf16(false, a, false, b, (short)0, c,
                                                 false, false);
}

// ---------------------------------------------------------------------------
// Elementwise fp32 -> bf16 convert, and fp32 [K][N] -> bf16 transposed [N][K]
// ---------------------------------------------------------------------------
__global__ void cvt_kernel(const float* __restrict__ src, bf16_t* __restrict__ dst,
                           int n) {
  int i = blockIdx.x * blockDim.x + threadIdx.x;
  if (i < n) dst[i] = (bf16_t)src[i];
}

__global__ void cvt_t_kernel(const float* __restrict__ src, bf16_t* __restrict__ dst,
                             int K, int N) {
  int i = blockIdx.x * blockDim.x + threadIdx.x;
  if (i >= K * N) return;
  int n = i % N, k = i / N;
  dst[(size_t)n * K + k] = (bf16_t)src[i];
}

// ---------------------------------------------------------------------------
// TDM descriptor helper: stage a [128 rows x 32 k] bf16 tile (row stride K
// elements in global) into LDS with 16B padding after every 64B row
// (LDS row stride 80B -> 20-bank stride, conflict-free fragment reads).
// D# layout per CDNA5 ISA.
// ---------------------------------------------------------------------------
__device__ __forceinline__ void tdm_stage_tile(const bf16_t* gbase, size_t row0,
                                               int k0, int K, unsigned lds_off) {
  unsigned long long ga =
      (unsigned long long)(size_t)gbase + (row0 * (size_t)K + (size_t)k0) * 2ull;
  // group0: count=1 | lds_addr | global_addr | type=2
  u32x4 g0 = {1u, lds_off, (unsigned)ga,
              (unsigned)(ga >> 32) | (2u << 30)};
  // group1: data_size=1(2B) | pad_enable | pad_interval=3 (64B) | pad_amount=3 (16B)
  const unsigned TD = 1u << 20;  // generous tensor dims (tiles never clip)
  i32x8 g1 = {
      (int)0x06D10000u,                              // mask|ds|pad cfg
      (int)((TD & 0xFFFFu) << 16),                   // atomic_addr=0 | dim0 lo
      (int)((TD >> 16) | ((TD & 0xFFFFu) << 16)),    // dim0 hi | dim1 lo
      (int)((TD >> 16) | (32u << 16)),               // dim1 hi | tile_dim0=32
      (int)128,                                      // tile_dim1=128 | tile_dim2=0
      (int)K,                                        // tensor_dim0_stride lo
      0, 0};                                         // stride hi | dim1_stride
  TDM_LOAD(g0, g1);
}

// ---------------------------------------------------------------------------
// GEMM: C[M][N] = A[M][K] * Bt[N][K]^T, bf16 in, f32 accumulate.
// Workgroup (256 thr = 8 waves) computes a 128x128 tile; wave (wm,wn) owns a
// 32x64 sub-tile as 2x4 accumulators.  A/B tiles are staged to LDS by the
// Tensor Data Mover (wave 0 issues, ping-pong buffers overlap DMA w/ WMMA).
// LDS row stride = 40 bf16 (80B) due to TDM padding.
// ---------------------------------------------------------------------------
#define LROW 40
__global__ void gemm_wmma_tdm(const bf16_t* __restrict__ A,
                              const bf16_t* __restrict__ Bt,
                              void* __restrict__ C, int M, int N, int K,
                              int storeF32) {
  __shared__ __align__(16) bf16_t ldsA[2][128 * LROW];
  __shared__ __align__(16) bf16_t ldsB[2][128 * LROW];

  const int lane = threadIdx.x & 31;
  const int col  = lane & 15;
  const int hi   = lane >> 4;
  const int w    = threadIdx.x >> 5;
  const int wm   = w >> 1;       // 0..3  (M sub-tile of 32)
  const int wn   = w & 1;        // 0..1  (N sub-tile of 64)

  const int tilesN = N >> 7;
  const int wgm = blockIdx.x / tilesN;
  const int wgn = blockIdx.x % tilesN;
  const size_t rowA0 = (size_t)wgm * 128;
  const size_t rowB0 = (size_t)wgn * 128;

  v8f acc[2][4];
#pragma unroll
  for (int i = 0; i < 2; ++i)
#pragma unroll
    for (int j = 0; j < 4; ++j) acc[i][j] = (v8f){};

  const int nkb = K >> 5;
  // preload first k-tile into buffer 0
  if (w == 0) {
    tdm_stage_tile(A, rowA0, 0, K, (unsigned)(size_t)&ldsA[0][0]);
    tdm_stage_tile(Bt, rowB0, 0, K, (unsigned)(size_t)&ldsB[0][0]);
  }

  for (int kb = 0; kb < nkb; ++kb) {
    const int cur = kb & 1;
    __syncthreads();  // everyone done reading buffer cur^1 from iter kb-1
    if (w == 0) {
      if (kb + 1 < nkb) {
        tdm_stage_tile(A, rowA0, (kb + 1) * 32, K,
                       (unsigned)(size_t)&ldsA[cur ^ 1][0]);
        tdm_stage_tile(Bt, rowB0, (kb + 1) * 32, K,
                       (unsigned)(size_t)&ldsB[cur ^ 1][0]);
        __builtin_amdgcn_s_wait_tensorcnt(2);  // in-order: current tile landed
      } else {
        __builtin_amdgcn_s_wait_tensorcnt(0);
      }
    }
    __syncthreads();  // current tile visible to all waves

    v16bf a0 = load_frag(&ldsA[cur][(wm * 32 + col) * LROW], 0, hi);
    v16bf a1 = load_frag(&ldsA[cur][(wm * 32 + 16 + col) * LROW], 0, hi);
#pragma unroll
    for (int j = 0; j < 4; ++j) {
      v16bf bj = load_frag(&ldsB[cur][(wn * 64 + 16 * j + col) * LROW], 0, hi);
      acc[0][j] = wmma_bf16(a0, bj, acc[0][j]);
      acc[1][j] = wmma_bf16(a1, bj, acc[1][j]);
    }
  }

  // C/D layout: element r <-> local row +r+8*hi
#pragma unroll
  for (int i = 0; i < 2; ++i)
#pragma unroll
    for (int j = 0; j < 4; ++j)
#pragma unroll
      for (int r = 0; r < 8; ++r) {
        size_t row = rowA0 + wm * 32 + 16 * i + r + 8 * hi;
        size_t cc  = rowB0 + wn * 64 + 16 * j + col;
        size_t idx = row * N + cc;
        if (storeF32) ((float*)C)[idx] = acc[i][j][r];
        else          ((bf16_t*)C)[idx] = (bf16_t)acc[i][j][r];
      }
}

// ---------------------------------------------------------------------------
// RoPE + rearrange:
//   Qraw [B*L][H*DH]  -> Qr [B][H][L][DH]   (rope applied)
//   Kraw [B*L][KV*DH] -> Kr [B][KV][L][DH]  (rope applied)
//   Vraw [B*L][KV*DH] -> Vt [B][KV][DH][L]  (transposed for AV B-fragments)
// ---------------------------------------------------------------------------
__global__ void rope_rearrange(const bf16_t* __restrict__ Qraw,
                               const bf16_t* __restrict__ Kraw,
                               const bf16_t* __restrict__ Vraw,
                               const float* __restrict__ cosb,
                               const float* __restrict__ sinb,
                               bf16_t* __restrict__ Qr, bf16_t* __restrict__ Kr,
                               bf16_t* __restrict__ Vt) {
  size_t idx = (size_t)blockIdx.x * blockDim.x + threadIdx.x;
  if (idx >= (size_t)BB * LL * HH * DH) return;
  int d   = idx & (DH - 1);
  size_t t = idx >> 6;
  int h   = t & (HH - 1); t >>= 4;
  int pos = t & (LL - 1); t >>= 11;
  int b   = (int)t;

  int dh = d & 31;
  float c = cosb[pos * 32 + dh];
  float s = sinb[pos * 32 + dh];

  size_t qin = ((size_t)(b * LL + pos)) * (HH * DH) + h * DH;
  float x1 = (float)Qraw[qin + dh];
  float x2 = (float)Qraw[qin + 32 + dh];
  float qv = (d < 32) ? (x1 * c - x2 * s) : (x2 * c + x1 * s);
  Qr[(((size_t)b * HH + h) * LL + pos) * DH + d] = (bf16_t)qv;

  if (h < KVH) {
    size_t kin = ((size_t)(b * LL + pos)) * (KVH * DH) + h * DH;
    float k1 = (float)Kraw[kin + dh];
    float k2 = (float)Kraw[kin + 32 + dh];
    float kv = (d < 32) ? (k1 * c - k2 * s) : (k2 * c + k1 * s);
    Kr[(((size_t)b * KVH + h) * LL + pos) * DH + d] = (bf16_t)kv;
    Vt[(((size_t)b * KVH + h) * DH + d) * LL + pos] = Vraw[kin + d];
  }
}

// ---------------------------------------------------------------------------
// Flash-style causal GQA attention.  One wave per (b,h,16-query tile); key
// blocks of 32: 4 WMMAs for S, online softmax with wave32 shfl row
// reductions, LDS transpose of P into A-layout, 4 WMMAs for AV.
// ---------------------------------------------------------------------------
__global__ void attn_kernel(const bf16_t* __restrict__ Qr,
                            const bf16_t* __restrict__ Kr,
                            const bf16_t* __restrict__ Vt,
                            bf16_t* __restrict__ Oout) {
  __shared__ __align__(16) bf16_t Pl[4][16][32];

  const int lane = threadIdx.x & 31;
  const int col  = lane & 15;
  const int hi   = lane >> 4;
  const int w    = threadIdx.x >> 5;
  const int gw   = blockIdx.x * 4 + w;

  const int head = gw >> 7;
  const int qt   = gw & 127;
  const int b    = head >> 4;
  const int h    = head & (HH - 1);
  const int kv   = h >> 2;
  const int q0   = qt * 16;

  const bf16_t* Qh = Qr + (((size_t)b * HH + h) * LL) * DH;
  const bf16_t* Kh = Kr + (((size_t)b * KVH + kv) * LL) * DH;
  const bf16_t* Vh = Vt + (((size_t)b * KVH + kv) * DH) * LL;

  const bf16_t* qrow = Qh + (size_t)(q0 + col) * DH;
  v16bf qA0 = load_frag(qrow, 0, hi);
  v16bf qA1 = load_frag(qrow, 32, hi);

  float m[8], l[8];
  v8f o[4];
#pragma unroll
  for (int r = 0; r < 8; ++r) { m[r] = -__builtin_inff(); l[r] = 0.f; }
#pragma unroll
  for (int n = 0; n < 4; ++n) o[n] = (v8f){};

  const float scale = 0.125f;
  const int nkb = (q0 + 16 + 31) >> 5;

  for (int kb = 0; kb < nkb; ++kb) {
    const int j0 = kb * 32;

    v8f S0 = {}, S1 = {};
    {
      const bf16_t* krow0 = Kh + (size_t)(j0 + col) * DH;
      const bf16_t* krow1 = Kh + (size_t)(j0 + 16 + col) * DH;
      S0 = wmma_bf16(qA0, load_frag(krow0, 0, hi), S0);
      S0 = wmma_bf16(qA1, load_frag(krow0, 32, hi), S0);
      S1 = wmma_bf16(qA0, load_frag(krow1, 0, hi), S1);
      S1 = wmma_bf16(qA1, load_frag(krow1, 32, hi), S1);
    }

    const int key0 = j0 + col;
    const int key1 = j0 + 16 + col;
#pragma unroll
    for (int r = 0; r < 8; ++r) {
      const int q = q0 + r + 8 * hi;
      S0[r] = (key0 <= q) ? S0[r] * scale : -3.0e38f;
      S1[r] = (key1 <= q) ? S1[r] * scale : -3.0e38f;
    }

#pragma unroll
    for (int r = 0; r < 8; ++r) {
      float mx = fmaxf(S0[r], S1[r]);
      mx = fmaxf(mx, __shfl_xor(mx, 1, 32));
      mx = fmaxf(mx, __shfl_xor(mx, 2, 32));
      mx = fmaxf(mx, __shfl_xor(mx, 4, 32));
      mx = fmaxf(mx, __shfl_xor(mx, 8, 32));
      const float nm = fmaxf(m[r], mx);
      const float alpha = __expf(m[r] - nm);
      S0[r] = __expf(S0[r] - nm);
      S1[r] = __expf(S1[r] - nm);
      float sum = S0[r] + S1[r];
      sum += __shfl_xor(sum, 1, 32);
      sum += __shfl_xor(sum, 2, 32);
      sum += __shfl_xor(sum, 4, 32);
      sum += __shfl_xor(sum, 8, 32);
      l[r] = l[r] * alpha + sum;
      m[r] = nm;
      o[0][r] *= alpha; o[1][r] *= alpha; o[2][r] *= alpha; o[3][r] *= alpha;
    }

#pragma unroll
    for (int r = 0; r < 8; ++r) {
      Pl[w][r + 8 * hi][col]      = (bf16_t)S0[r];
      Pl[w][r + 8 * hi][16 + col] = (bf16_t)S1[r];
    }
    v16bf pA = load_frag(&Pl[w][col][0], 0, hi);

#pragma unroll
    for (int n = 0; n < 4; ++n) {
      const bf16_t* vrow = Vh + (size_t)(16 * n + col) * LL;
      v16bf vB = load_frag(vrow + j0, 0, hi);
      o[n] = wmma_bf16(pA, vB, o[n]);
    }
  }

#pragma unroll
  for (int n = 0; n < 4; ++n) {
#pragma unroll
    for (int r = 0; r < 8; ++r) {
      const int q = q0 + r + 8 * hi;
      Oout[((size_t)(b * LL + q)) * (HH * DH) + h * DH + 16 * n + col] =
          (bf16_t)(o[n][r] / l[r]);
    }
  }
}

// ---------------------------------------------------------------------------
extern "C" void kernel_launch(void* const* d_in, const int* in_sizes, int n_in,
                              void* d_out, int out_size, void* d_ws, size_t ws_size,
                              hipStream_t stream) {
  (void)in_sizes; (void)n_in; (void)out_size; (void)ws_size;
  const float* x  = (const float*)d_in[0];
  const float* rc = (const float*)d_in[1];
  const float* rs = (const float*)d_in[2];
  const float* Wq = (const float*)d_in[3];
  const float* Wk = (const float*)d_in[4];
  const float* Wv = (const float*)d_in[5];
  const float* Wo = (const float*)d_in[6];

  char* ws = (char*)d_ws;
  size_t off = 0;
  auto alloc = [&](size_t bytes) -> char* {
    char* p = ws + off;
    off = (off + bytes + 255) & ~(size_t)255;
    return p;
  };

  const size_t ML = (size_t)BB * LL;  // 4096
  bf16_t* xb   = (bf16_t*)alloc(ML * DD * 2);
  bf16_t* Wqt  = (bf16_t*)alloc((size_t)DD * DD * 2);
  bf16_t* Wkt  = (bf16_t*)alloc((size_t)KVH * DH * DD * 2);
  bf16_t* Wvt  = (bf16_t*)alloc((size_t)KVH * DH * DD * 2);
  bf16_t* Wot  = (bf16_t*)alloc((size_t)DD * DD * 2);
  bf16_t* Qraw = (bf16_t*)alloc(ML * HH * DH * 2);
  bf16_t* Kraw = (bf16_t*)alloc(ML * KVH * DH * 2);
  bf16_t* Vraw = (bf16_t*)alloc(ML * KVH * DH * 2);
  bf16_t* Qr   = (bf16_t*)alloc(ML * HH * DH * 2);
  bf16_t* Kr   = (bf16_t*)alloc(ML * KVH * DH * 2);
  bf16_t* Vt   = (bf16_t*)alloc(ML * KVH * DH * 2);
  bf16_t* Oatt = (bf16_t*)alloc(ML * HH * DH * 2);

  {
    int n = (int)(ML * DD);
    cvt_kernel<<<(n + 255) / 256, 256, 0, stream>>>(x, xb, n);
    cvt_t_kernel<<<(DD * DD + 255) / 256, 256, 0, stream>>>(Wq, Wqt, DD, DD);
    cvt_t_kernel<<<(DD * KVH * DH + 255) / 256, 256, 0, stream>>>(Wk, Wkt, DD, KVH * DH);
    cvt_t_kernel<<<(DD * KVH * DH + 255) / 256, 256, 0, stream>>>(Wv, Wvt, DD, KVH * DH);
    cvt_t_kernel<<<(DD * DD + 255) / 256, 256, 0, stream>>>(Wo, Wot, DD, DD);
  }

  // QKV projection GEMMs (bf16 out): workgroup tiles of 128x128
  gemm_wmma_tdm<<<(int)(ML / 128) * (DD / 128), 256, 0, stream>>>(
      xb, Wqt, Qraw, (int)ML, DD, DD, 0);
  gemm_wmma_tdm<<<(int)(ML / 128) * ((KVH * DH) / 128), 256, 0, stream>>>(
      xb, Wkt, Kraw, (int)ML, KVH * DH, DD, 0);
  gemm_wmma_tdm<<<(int)(ML / 128) * ((KVH * DH) / 128), 256, 0, stream>>>(
      xb, Wvt, Vraw, (int)ML, KVH * DH, DD, 0);

  {
    size_t n = (size_t)BB * LL * HH * DH;
    rope_rearrange<<<(unsigned)((n + 255) / 256), 256, 0, stream>>>(
        Qraw, Kraw, Vraw, rc, rs, Qr, Kr, Vt);
  }

  attn_kernel<<<1024, 128, 0, stream>>>(Qr, Kr, Vt, Oatt);

  gemm_wmma_tdm<<<(int)(ML / 128) * (DD / 128), 256, 0, stream>>>(
      Oatt, Wot, d_out, (int)ML, DD, DD, 1);
}
